// GVPMultiStageModel_455266533900
// MI455X (gfx1250) — matrix operand order repaired and checked by reading, not compiled, not run on hardware
//
#include <hip/hip_runtime.h>
#include <stdint.h>

// ---------------------------------------------------------------------------
// CDNA5 (gfx1250) implementation of the GVP multi-stage model.
// All matmuls go through v_wmma_f32_16x16x32_bf16 (wave32 WMMA).
// Each wave computes a 16x(16*NT) strip: A fragment loaded once per k-step,
// NT independent WMMAs in flight (hides XDL latency, 1/NT the A traffic).
// ---------------------------------------------------------------------------

typedef __bf16 bf16_t;
typedef __attribute__((ext_vector_type(16))) __bf16 v16bf;
typedef __attribute__((ext_vector_type(8)))  float v8f;

static __device__ __forceinline__ bf16_t f2bf(float f) {
  union { float f; uint32_t u; } v; v.f = f;
  uint32_t r = v.u + 0x7fffu + ((v.u >> 16) & 1u);
  uint16_t h = (uint16_t)(r >> 16);
  bf16_t o; __builtin_memcpy(&o, &h, 2); return o;
}

// ------------------------------ kernels ------------------------------------

// Weight [K,N] f32 row-major -> transposed bf16 [Np, Kp], zero padded.
__global__ void k_wt(const float* __restrict__ W, bf16_t* __restrict__ Wt,
                     int K, int N, int Kp, int Np) {
  int idx = blockIdx.x * 256 + threadIdx.x;
  if (idx >= Kp * Np) return;
  int n = idx / Kp, k = idx - n * Kp;
  float v = (n < N && k < K) ? W[(size_t)k * N + n] : 0.f;
  Wt[idx] = f2bf(v);
}

// f32 [rows, lds] -> bf16 [rows, ldd], copy 'cols' columns, zero pad rest.
__global__ void k_cvt_rows(const float* __restrict__ src, int lds,
                           bf16_t* __restrict__ dst, int ldd, int total, int cols) {
  int idx = blockIdx.x * 256 + threadIdx.x;
  if (idx >= total) return;
  int r = idx / ldd, c = idx - r * ldd;
  float v = (c < cols) ? src[(size_t)r * lds + c] : 0.f;
  dst[idx] = f2bf(v);
}

// C[Mr,Np] (f32) = A[Mr,Kp](bf16) * Wt[Np,Kp](bf16)^T  (+bias, +relu)
// Wave computes 16 x (16*NT). K-loop of 32 via v_wmma_f32_16x16x32_bf16.
// A fragment: lane m=lane&15, slots e<8 -> K=k0+8*half+e ; e>=8 -> K=k0+8*half+8+e
// B fragment: lane n=lane&15, slots e<8 -> K=k0+8*half+e ; e>=8 -> K=k0+8*half+8+e
template <int NT>
__global__ __launch_bounds__(256) void k_gemm(
    const bf16_t* __restrict__ A, const bf16_t* __restrict__ Wt,
    const float* __restrict__ bias, float* __restrict__ C,
    int Mr, int Np, int Kp, int Ntrue, int relu) {
  int wave = threadIdx.x >> 5;
  int lane = threadIdx.x & 31;
  int hv   = lane >> 4;
  int l15  = lane & 15;
  int row0 = (blockIdx.y * 8 + wave) * 16;
  if (row0 >= Mr) return;
  int col0 = blockIdx.x * (16 * NT);
  const bf16_t* arow  = A  + (size_t)(row0 + l15) * Kp;
  const bf16_t* brow0 = Wt + (size_t)(col0 + l15) * Kp;
  v8f acc[NT] = {};
  for (int k0 = 0; k0 < Kp; k0 += 32) {
    union { uint4 q[2]; v16bf v; } ua;
    const bf16_t* ap = arow + k0 + 8 * hv;
    ua.q[0] = *(const uint4*)ap;
    ua.q[1] = *(const uint4*)(ap + 16);
#pragma unroll
    for (int t = 0; t < NT; ++t) {
      union { uint4 q[2]; v16bf v; } ub;
      const bf16_t* bp = brow0 + (size_t)t * 16 * Kp + k0 + 8 * hv;
      ub.q[0] = *(const uint4*)bp;
      ub.q[1] = *(const uint4*)(bp + 16);
      acc[t] = __builtin_amdgcn_wmma_f32_16x16x32_bf16(false, ua.v, false, ub.v,
                                                       (short)0, acc[t], false, false);
    }
  }
#pragma unroll
  for (int t = 0; t < NT; ++t) {
    int col = col0 + t * 16 + l15;
    float bv = (bias && col < Ntrue) ? bias[col] : 0.f;
    float* cp = C + (size_t)(row0 + 8 * hv) * Np + col;
#pragma unroll
    for (int r = 0; r < 8; ++r) {
      float v = acc[t][r] + bv;
      if (relu) v = fmaxf(v, 0.f);
      cp[(size_t)r * Np] = v;
    }
  }
}

// LayerNorm over S cols, one wave per row (wave32 shuffle reduction).
__global__ void k_ln_s(const float* __restrict__ X, float* __restrict__ Y,
                       const float* __restrict__ g, const float* __restrict__ b,
                       int M, int ld, int S) {
  int row = blockIdx.x * 8 + (threadIdx.x >> 5);
  if (row >= M) return;
  int lane = threadIdx.x & 31;
  const float* x = X + (size_t)row * ld;
  float s = 0.f, q = 0.f;
  for (int c = lane; c < S; c += 32) { float v = x[c]; s += v; q += v * v; }
  for (int o = 16; o; o >>= 1) { s += __shfl_xor(s, o, 32); q += __shfl_xor(q, o, 32); }
  float mu = s / S, var = q / S - mu * mu;
  float inv = rsqrtf(var + 1e-5f);
  float* y = Y + (size_t)row * ld;
  for (int c = lane; c < S; c += 32) y[c] = (x[c] - mu) * inv * g[c] + b[c];
}

// Vector LN on planar V [(n*3+d)*ld + k]: V /= sqrt(mean_k clip(|v_k|^2,eps))
__global__ void k_ln_v(float* __restrict__ V, int ld, int N, int nv) {
  int n = blockIdx.x * 256 + threadIdx.x;
  if (n >= N) return;
  float* p = V + (size_t)n * 3 * ld;
  float acc = 0.f;
  for (int k = 0; k < nv; ++k) {
    float a = p[k], b = p[ld + k], c = p[2 * ld + k];
    acc += fmaxf(a * a + b * b + c * c, 1e-8f);
  }
  float inv = rsqrtf(acc / nv);
  for (int k = 0; k < nv; ++k) { p[k] *= inv; p[ld + k] *= inv; p[2 * ld + k] *= inv; }
}

// vn[i,h] = sqrt(clip(sum_d Vh[(i*3+d),h]^2, eps)); vn stride fixed 64.
__global__ void k_vn(const float* __restrict__ Vh, int ld, float* __restrict__ vn,
                     int M, int H) {
  int idx = blockIdx.x * 256 + threadIdx.x;
  if (idx >= M * H) return;
  int i = idx / H, h = idx - i * H;
  const float* p = Vh + (size_t)i * 3 * ld + h;
  float a = p[0], b = p[ld], c = p[2 * ld];
  vn[(size_t)i * 64 + h] = sqrtf(fmaxf(a * a + b * b + c * c, 1e-8f));
}

// V *= sigmoid(||V[i,:,h]||) over the 3 spatial comps (planar layout).
__global__ void k_gate(float* __restrict__ V, int ld, int M, int H) {
  int idx = blockIdx.x * 256 + threadIdx.x;
  if (idx >= M * H) return;
  int i = idx / H, h = idx - i * H;
  float* p = V + (size_t)i * 3 * ld + h;
  float a = p[0], b = p[ld], c = p[2 * ld];
  float nrm = sqrtf(fmaxf(a * a + b * b + c * c, 1e-8f));
  float g = 1.f / (1.f + expf(-nrm));
  p[0] = a * g; p[ld] = b * g; p[2 * ld] = c * g;
}

// out(bf16)[r, 0..S-1]=s ; [S..S+H-1]=vn ; rest 0. Rows >= M -> 0.
__global__ void k_concat_svn(const float* __restrict__ s, int lds, int S,
                             const float* __restrict__ vn, int vnld, int H,
                             bf16_t* __restrict__ out, int ldd, int M, int total) {
  int idx = blockIdx.x * 256 + threadIdx.x;
  if (idx >= total) return;
  int r = idx / ldd, c = idx - r * ldd;
  float v = 0.f;
  if (r < M) {
    if (c < S) v = s[(size_t)r * lds + c];
    else if (c < S + H) v = vn[(size_t)r * vnld + (c - S)];
  }
  out[idx] = f2bf(v);
}

// node_v [N,3(vec),3(spatial)] -> planar [(n*3+d)*32 + k], k<3 valid
__global__ void k_vplanar(const float* __restrict__ nv3, float* __restrict__ Vp,
                          int N, int total) {
  int idx = blockIdx.x * 256 + threadIdx.x;
  if (idx >= total) return;
  int r = idx >> 5, k = idx & 31;
  float v = 0.f;
  if (r < 3 * N && k < 3) { int n = r / 3, d = r - 3 * n; v = nv3[(size_t)n * 9 + k * 3 + d]; }
  Vp[idx] = v;
}

// s0 = [node_s(6) | seq_emb[seq](20) | 0...]
__global__ void k_build_seq(const float* __restrict__ ns, const int* __restrict__ seq,
                            const float* __restrict__ emb, float* __restrict__ s0,
                            int ld, int N, int total) {
  int idx = blockIdx.x * 256 + threadIdx.x;
  if (idx >= total) return;
  int r = idx / ld, c = idx - r * ld;
  float v = 0.f;
  if (r < N) {
    if (c < 6) v = ns[(size_t)r * 6 + c];
    else if (c < 26) v = emb[(size_t)seq[r] * 20 + (c - 6)];
  }
  s0[idx] = v;
}

// We (edge GVP, vi=vo=h=1): LN on eV then Vh=eVln*wh ; vn=|Vh| ; eVo=Vh*wv
__global__ void k_we(const float* __restrict__ eV, const float* __restrict__ wh,
                     const float* __restrict__ wv, float* __restrict__ eVo,
                     float* __restrict__ vnWe, int E) {
  int e = blockIdx.x * 256 + threadIdx.x;
  if (e >= E) return;
  float x = eV[(size_t)e * 3], y = eV[(size_t)e * 3 + 1], z = eV[(size_t)e * 3 + 2];
  float inv = rsqrtf(fmaxf(x * x + y * y + z * z, 1e-8f));
  float whv = wh[0], wvv = wv[0];
  float vx = x * inv * whv, vy = y * inv * whv, vz = z * inv * whv;
  vnWe[e] = sqrtf(fmaxf(vx * vx + vy * vy + vz * vz, 1e-8f));
  eVo[(size_t)e * 3] = vx * wvv; eVo[(size_t)e * 3 + 1] = vy * wvv; eVo[(size_t)e * 3 + 2] = vz * wvv;
}

// Edge vector message gather -> bf16 A [(i*3+d), 64]: [V[src](16)|eVo(1)|V[dst](16)|0]
__global__ void k_gmv(const float* __restrict__ Vp, const float* __restrict__ eVo,
                      const int* __restrict__ src, const int* __restrict__ dst,
                      int e0, int n, bf16_t* __restrict__ out, int total) {
  int idx = blockIdx.x * 256 + threadIdx.x;
  if (idx >= total) return;
  int r = idx >> 6, k = idx & 63;
  int i = r / 3, d = r - 3 * i;
  float v = 0.f;
  if (i < n) {
    int e = e0 + i;
    if (k < 16)       v = Vp[((size_t)src[e] * 3 + d) * 32 + k];
    else if (k == 16) v = eVo[(size_t)e * 3 + d];
    else if (k < 33)  v = Vp[((size_t)dst[e] * 3 + d) * 32 + (k - 17)];
  }
  out[idx] = f2bf(v);
}

// Edge scalar message gather -> bf16 A [i, 352]: [s[src](128)|es(32)|s[dst](128)|vn(33)|0]
__global__ void k_gms(const float* __restrict__ s, const float* __restrict__ esP,
                      const float* __restrict__ vn, const int* __restrict__ src,
                      const int* __restrict__ dst, int e0, int n,
                      bf16_t* __restrict__ out, int total) {
  int idx = blockIdx.x * 256 + threadIdx.x;
  if (idx >= total) return;
  int r = idx / 352, c = idx - r * 352;
  float v = 0.f;
  if (r < n) {
    int e = e0 + r;
    if (c < 128)      v = s[(size_t)src[e] * 128 + c];
    else if (c < 160) v = esP[(size_t)e * 32 + (c - 128)];
    else if (c < 288) v = s[(size_t)dst[e] * 128 + (c - 160)];
    else if (c < 321) v = vn[(size_t)r * 64 + (c - 288)];
  }
  out[idx] = f2bf(v);
}

__global__ void k_degree(const int* __restrict__ dst, int E, float* __restrict__ deg) {
  int i = blockIdx.x * 256 + threadIdx.x;
  if (i >= E) return;
  atomicAdd(&deg[dst[i]], 1.f);
}

__global__ void k_seg_add_s(const float* __restrict__ ms, const int* __restrict__ dst,
                            int e0, int n, float* __restrict__ acc) {
  int idx = blockIdx.x * 256 + threadIdx.x;
  if (idx >= n * 128) return;
  int i = idx >> 7, c = idx & 127;
  atomicAdd(&acc[(size_t)dst[e0 + i] * 128 + c], ms[idx]);
}

__global__ void k_seg_add_v(const float* __restrict__ Vo, const int* __restrict__ dst,
                            int e0, int n, float* __restrict__ acc) {
  int idx = blockIdx.x * 256 + threadIdx.x;
  if (idx >= n * 48) return;
  int i = idx / 48, r = idx - 48 * i;
  int d = r >> 4, k = r & 15;
  atomicAdd(&acc[((size_t)dst[e0 + i] * 3 + d) * 32 + k],
            Vo[((size_t)i * 3 + d) * 32 + k]);
}

__global__ void k_resid_s(float* __restrict__ s, const float* __restrict__ acc,
                          const float* __restrict__ deg, int total) {
  int idx = blockIdx.x * 256 + threadIdx.x;
  if (idx >= total) return;
  int r = idx >> 7;
  s[idx] += acc[idx] / fmaxf(deg[r], 1.f);
}

__global__ void k_resid_v(float* __restrict__ V, const float* __restrict__ acc,
                          const float* __restrict__ deg, int total) {
  int idx = blockIdx.x * 256 + threadIdx.x;
  if (idx >= total) return;
  int r = idx >> 5; int n = r / 3;
  V[idx] += acc[idx] / fmaxf(deg[n], 1.f);
}

__global__ void k_add(float* __restrict__ a, const float* __restrict__ b, int n) {
  int i = blockIdx.x * 256 + threadIdx.x;
  if (i < n) a[i] += b[i];
}

__global__ void k_copy_cols(const float* __restrict__ src, int lds,
                            float* __restrict__ dst, int ldd, int off,
                            int cols, int total) {
  int idx = blockIdx.x * 256 + threadIdx.x;
  if (idx >= total) return;
  int r = idx / cols, c = idx - r * cols;
  dst[(size_t)r * ldd + off + c] = src[(size_t)r * lds + c];
}

// logits = h @ w2 + b2 + 0.5, one wave per row.
__global__ void k_dot_head(const float* __restrict__ H, const float* __restrict__ w2,
                           const float* __restrict__ b2, float* __restrict__ out,
                           int M, int K) {
  int row = blockIdx.x * 8 + (threadIdx.x >> 5);
  if (row >= M) return;
  int lane = threadIdx.x & 31;
  const float* h = H + (size_t)row * K;
  float s = 0.f;
  for (int k = lane; k < K; k += 32) s += h[k] * w2[k];
  for (int o = 16; o; o >>= 1) s += __shfl_xor(s, o, 32);
  if (!lane) out[row] = s + b2[0] + 0.5f;
}

__global__ void k_seg_graph(const float* __restrict__ lg, const int* __restrict__ n2g,
                            float* __restrict__ acc, float* __restrict__ cnt, int M) {
  int i = blockIdx.x * 256 + threadIdx.x;
  if (i >= M) return;
  int g = n2g[i];
  atomicAdd(&acc[g], lg[i]);
  atomicAdd(&cnt[g], 1.f);
}

__global__ void k_gfinal(const float* __restrict__ acc, const float* __restrict__ cnt,
                         float* __restrict__ out) {
  int b = threadIdx.x;
  if (b < 8) out[b] = acc[b] / fmaxf(cnt[b], 1.f);
}

// ------------------------------ host side ----------------------------------

static inline int rnd16(int x) { return (x + 15) & ~15; }
static inline int rnd32(int x) { return (x + 31) & ~31; }
static inline int gblk(int n) { return (n + 255) / 256; }

struct Arena { char* p; size_t off, cap; };
static float* allocF(Arena& a, size_t n) {
  size_t o = (a.off + 255) & ~(size_t)255; a.off = o + n * 4; return (float*)(a.p + o);
}
static bf16_t* allocB(Arena& a, size_t n) {
  size_t o = (a.off + 255) & ~(size_t)255; a.off = o + n * 2; return (bf16_t*)(a.p + o);
}

struct WMat { bf16_t* t; int Kp, Np; };
static WMat prepW(hipStream_t st, Arena& ar, const float* W, int K, int N) {
  WMat m; m.Kp = rnd32(K); m.Np = rnd32(N);
  m.t = allocB(ar, (size_t)m.Kp * m.Np);
  int tot = m.Kp * m.Np;
  k_wt<<<gblk(tot), 256, 0, st>>>(W, m.t, K, N, m.Kp, m.Np);
  return m;
}
static void gemm(hipStream_t st, const bf16_t* A, const WMat& W, const float* bias,
                 float* C, int Mr, int Ntrue, int relu) {
  dim3 blk(256);
  if (W.Np % 64 == 0) {
    dim3 g(W.Np / 64, (Mr + 127) / 128);
    k_gemm<4><<<g, blk, 0, st>>>(A, W.t, bias, C, Mr, W.Np, W.Kp, Ntrue, relu);
  } else {
    dim3 g(W.Np / 32, (Mr + 127) / 128);
    k_gemm<2><<<g, blk, 0, st>>>(A, W.t, bias, C, Mr, W.Np, W.Kp, Ntrue, relu);
  }
}

struct LNW { const float *b, *g; };
struct GVPW { const float *wh, *wsb, *wsw, *wv; WMat whT, wswT, wvT;
              int vi, h, si, so, vo; };
struct LayerW { GVPW conv[3]; GVPW ff[2]; LNW n0, n1; };
struct StageW { GVPW We; LNW We_ln; GVPW Wout; LNW Wout_ln; GVPW Wv; LNW Wv_ln;
                LayerW L[3]; const float* seq; };
struct DenseW { const float *b1, *b2, *w1, *w2; };

// Params arrive either as separate leaves (jax tree order: sorted dict keys)
// or as one flat blob; Rd handles both.
struct Rd {
  void* const* din; int i; const float* blob; size_t off; bool blobMode;
  const float* take(size_t n) {
    if (blobMode) { const float* p = blob + off; off += n; return p; }
    (void)n; return (const float*)din[i++];
  }
};
static void rdGVP(Rd& r, GVPW& g, int vi, int h, int si, int so, int vo) {
  g.vi = vi; g.h = h; g.si = si; g.so = so; g.vo = vo;
  g.wh  = r.take((size_t)vi * h);
  g.wsb = r.take((size_t)so);
  g.wsw = r.take((size_t)(si + h) * so);
  g.wv  = (vo > 0) ? r.take((size_t)h * vo) : nullptr;
}
static void rdLN(Rd& r, LNW& l, int sz) { l.b = r.take(sz); l.g = r.take(sz); }
static void rdStage(Rd& r, StageW& s, int si, bool seq) {
  rdGVP(r, s.We, 1, 1, 32, 32, 1);     rdLN(r, s.We_ln, 32);
  rdGVP(r, s.Wout, 48, 48, 384, 384, 0); rdLN(r, s.Wout_ln, 384);
  rdGVP(r, s.Wv, 3, 16, si, 128, 16);  rdLN(r, s.Wv_ln, si);
  for (int l = 0; l < 3; ++l) {
    rdGVP(r, s.L[l].conv[0], 33, 33, 288, 128, 16);
    rdGVP(r, s.L[l].conv[1], 16, 16, 128, 128, 16);
    rdGVP(r, s.L[l].conv[2], 16, 16, 128, 128, 16);
    rdGVP(r, s.L[l].ff[0], 16, 32, 128, 512, 32);
    rdGVP(r, s.L[l].ff[1], 32, 32, 512, 128, 16);
    rdLN(r, s.L[l].n0, 128); rdLN(r, s.L[l].n1, 128);
  }
  s.seq = seq ? r.take(400) : nullptr;
}
static void prepStage(hipStream_t st, Arena& ar, StageW& s) {
  s.We.wswT = prepW(st, ar, s.We.wsw, 33, 32);   // wh,wv (1x1) used raw
  s.Wv.whT  = prepW(st, ar, s.Wv.wh, 3, 16);
  s.Wv.wswT = prepW(st, ar, s.Wv.wsw, s.Wv.si + 16, 128);
  s.Wv.wvT  = prepW(st, ar, s.Wv.wv, 16, 16);
  s.Wout.whT  = prepW(st, ar, s.Wout.wh, 48, 48);
  s.Wout.wswT = prepW(st, ar, s.Wout.wsw, 432, 384);
  for (int l = 0; l < 3; ++l) {
    for (int c = 0; c < 3; ++c) {
      GVPW& g = s.L[l].conv[c];
      g.whT  = prepW(st, ar, g.wh, g.vi, g.h);
      g.wswT = prepW(st, ar, g.wsw, g.si + g.h, g.so);
      g.wvT  = prepW(st, ar, g.wv, g.h, g.vo);
    }
    for (int f = 0; f < 2; ++f) {
      GVPW& g = s.L[l].ff[f];
      g.whT  = prepW(st, ar, g.wh, g.vi, g.h);
      g.wswT = prepW(st, ar, g.wsw, g.si + g.h, g.so);
      g.wvT  = prepW(st, ar, g.wv, g.h, g.vo);
    }
  }
}

struct Bufs {
  float *sCur, *Vp, *sAcc, *VAcc, *cS, *cV, *esP, *eVo, *vnWe, *deg, *s0p, *s0c;
  float *F_VH, *F_VO, *F_VN, *F_MS, *gAcc;
  bf16_t *BF_AV, *BF_VH, *BF_AS;
};

static void runStage(hipStream_t st, const StageW& w, int N, int E,
                     const float* node_s, const float* node_v,
                     const float* edge_s, const float* edge_v,
                     const int* seq, const int* src, const int* dst,
                     int si, float* s0, int s0ld,
                     float* outDst, int outLd, const Bufs& B) {
  int Nr = rnd16(N), N3 = rnd16(3 * N), Er = rnd16(E);
  // ---- node vector features -> planar [N*3, 32]
  k_vplanar<<<gblk(N3 * 32), 256, 0, st>>>(node_v, B.Vp, N, N3 * 32);
  if (node_s)
    k_build_seq<<<gblk(Nr * s0ld), 256, 0, st>>>(node_s, seq, w.seq, s0, s0ld, N, Nr * s0ld);
  // ---- Wv_ln
  k_ln_s<<<(N + 7) / 8, 256, 0, st>>>(s0, s0, w.Wv_ln.g, w.Wv_ln.b, N, s0ld, si);
  k_ln_v<<<gblk(N), 256, 0, st>>>(B.Vp, 32, N, 3);
  // ---- Wv GVP (act=False)
  k_cvt_rows<<<gblk(N3 * 32), 256, 0, st>>>(B.Vp, 32, B.BF_AV, 32, N3 * 32, 3);
  gemm(st, B.BF_AV, w.Wv.whT, nullptr, B.F_VH, N3, 16, 0);
  k_vn<<<gblk(N * 16), 256, 0, st>>>(B.F_VH, 32, B.F_VN, N, 16);
  int aldv = rnd32(si + 16);
  k_concat_svn<<<gblk(Nr * aldv), 256, 0, st>>>(s0, s0ld, si, B.F_VN, 64, 16, B.BF_AS, aldv, N, Nr * aldv);
  gemm(st, B.BF_AS, w.Wv.wswT, w.Wv.wsb, B.sCur, Nr, 128, 0);
  k_cvt_rows<<<gblk(N3 * 32), 256, 0, st>>>(B.F_VH, 32, B.BF_VH, 32, N3 * 32, 16);
  gemm(st, B.BF_VH, w.Wv.wvT, nullptr, B.Vp, N3, 16, 0);
  // ---- We_ln + We on all edges
  k_ln_s<<<(E + 7) / 8, 256, 0, st>>>(edge_s, B.F_MS, w.We_ln.g, w.We_ln.b, E, 32, 32);
  k_we<<<gblk(E), 256, 0, st>>>(edge_v, w.We.wh, w.We.wv, B.eVo, B.vnWe, E);
  k_concat_svn<<<gblk(Er * 64), 256, 0, st>>>(B.F_MS, 32, 32, B.vnWe, 1, 1, B.BF_AS, 64, E, Er * 64);
  gemm(st, B.BF_AS, w.We.wswT, w.We.wsb, B.esP, Er, 32, 0);
  // ---- degree (shared by all layers)
  (void)hipMemsetAsync(B.deg, 0, (size_t)Nr * 4, st);
  k_degree<<<gblk(E), 256, 0, st>>>(dst, E, B.deg);

  const int CH = 32768;
  for (int l = 0; l < 3; ++l) {
    const GVPW &c0 = w.L[l].conv[0], &c1 = w.L[l].conv[1], &c2 = w.L[l].conv[2];
    (void)hipMemsetAsync(B.sAcc, 0, (size_t)Nr * 128 * 4, st);
    (void)hipMemsetAsync(B.VAcc, 0, (size_t)N3 * 32 * 4, st);
    for (int e0 = 0; e0 < E; e0 += CH) {
      int n = (E - e0 < CH) ? (E - e0) : CH;
      int nr = rnd16(n), n3 = rnd16(3 * n);
      // conv GVP0 (vi=33 -> vo=16), act
      k_gmv<<<gblk(n3 * 64), 256, 0, st>>>(B.Vp, B.eVo, src, dst, e0, n, B.BF_AV, n3 * 64);
      gemm(st, B.BF_AV, c0.whT, nullptr, B.F_VH, n3, 33, 0);
      k_vn<<<gblk(n * 33), 256, 0, st>>>(B.F_VH, 64, B.F_VN, n, 33);
      k_gms<<<gblk(nr * 352), 256, 0, st>>>(B.sCur, B.esP, B.F_VN, src, dst, e0, n, B.BF_AS, nr * 352);
      gemm(st, B.BF_AS, c0.wswT, c0.wsb, B.F_MS, nr, 128, 1);
      k_cvt_rows<<<gblk(n3 * 64), 256, 0, st>>>(B.F_VH, 64, B.BF_VH, 64, n3 * 64, 33);
      gemm(st, B.BF_VH, c0.wvT, nullptr, B.F_VO, n3, 16, 0);
      k_gate<<<gblk(n * 16), 256, 0, st>>>(B.F_VO, 32, n, 16);
      // conv GVP1, act
      k_cvt_rows<<<gblk(n3 * 32), 256, 0, st>>>(B.F_VO, 32, B.BF_AV, 32, n3 * 32, 16);
      gemm(st, B.BF_AV, c1.whT, nullptr, B.F_VH, n3, 16, 0);
      k_vn<<<gblk(n * 16), 256, 0, st>>>(B.F_VH, 32, B.F_VN, n, 16);
      k_concat_svn<<<gblk(nr * 160), 256, 0, st>>>(B.F_MS, 128, 128, B.F_VN, 64, 16, B.BF_AS, 160, n, nr * 160);
      gemm(st, B.BF_AS, c1.wswT, c1.wsb, B.F_MS, nr, 128, 1);
      k_cvt_rows<<<gblk(n3 * 32), 256, 0, st>>>(B.F_VH, 32, B.BF_VH, 32, n3 * 32, 16);
      gemm(st, B.BF_VH, c1.wvT, nullptr, B.F_VO, n3, 16, 0);
      k_gate<<<gblk(n * 16), 256, 0, st>>>(B.F_VO, 32, n, 16);
      // conv GVP2, no act
      k_cvt_rows<<<gblk(n3 * 32), 256, 0, st>>>(B.F_VO, 32, B.BF_AV, 32, n3 * 32, 16);
      gemm(st, B.BF_AV, c2.whT, nullptr, B.F_VH, n3, 16, 0);
      k_vn<<<gblk(n * 16), 256, 0, st>>>(B.F_VH, 32, B.F_VN, n, 16);
      k_concat_svn<<<gblk(nr * 160), 256, 0, st>>>(B.F_MS, 128, 128, B.F_VN, 64, 16, B.BF_AS, 160, n, nr * 160);
      gemm(st, B.BF_AS, c2.wswT, c2.wsb, B.F_MS, nr, 128, 0);
      k_cvt_rows<<<gblk(n3 * 32), 256, 0, st>>>(B.F_VH, 32, B.BF_VH, 32, n3 * 32, 16);
      gemm(st, B.BF_VH, c2.wvT, nullptr, B.F_VO, n3, 16, 0);
      // segment sum into dst nodes
      k_seg_add_s<<<gblk(n * 128), 256, 0, st>>>(B.F_MS, dst, e0, n, B.sAcc);
      k_seg_add_v<<<gblk(n * 48), 256, 0, st>>>(B.F_VO, dst, e0, n, B.VAcc);
    }
    // residual (mean) + norm0
    k_resid_s<<<gblk(N * 128), 256, 0, st>>>(B.sCur, B.sAcc, B.deg, N * 128);
    k_resid_v<<<gblk(3 * N * 32), 256, 0, st>>>(B.Vp, B.VAcc, B.deg, 3 * N * 32);
    k_ln_s<<<(N + 7) / 8, 256, 0, st>>>(B.sCur, B.sCur, w.L[l].n0.g, w.L[l].n0.b, N, 128, 128);
    k_ln_v<<<gblk(N), 256, 0, st>>>(B.Vp, 32, N, 16);
    // feed-forward: GVP(128,16 -> 512,32, act) then GVP(512,32 -> 128,16)
    const GVPW &f0 = w.L[l].ff[0], &f1 = w.L[l].ff[1];
    k_cvt_rows<<<gblk(N3 * 32), 256, 0, st>>>(B.Vp, 32, B.BF_AV, 32, N3 * 32, 16);
    gemm(st, B.BF_AV, f0.whT, nullptr, B.F_VH, N3, 32, 0);
    k_vn<<<gblk(N * 32), 256, 0, st>>>(B.F_VH, 32, B.F_VN, N, 32);
    k_concat_svn<<<gblk(Nr * 160), 256, 0, st>>>(B.sCur, 128, 128, B.F_VN, 64, 32, B.BF_AS, 160, N, Nr * 160);
    gemm(st, B.BF_AS, f0.wswT, f0.wsb, B.F_MS, Nr, 512, 1);
    k_cvt_rows<<<gblk(N3 * 32), 256, 0, st>>>(B.F_VH, 32, B.BF_VH, 32, N3 * 32, 32);
    gemm(st, B.BF_VH, f0.wvT, nullptr, B.F_VO, N3, 32, 0);
    k_gate<<<gblk(N * 32), 256, 0, st>>>(B.F_VO, 32, N, 32);
    k_cvt_rows<<<gblk(N3 * 32), 256, 0, st>>>(B.F_VO, 32, B.BF_AV, 32, N3 * 32, 32);
    gemm(st, B.BF_AV, f1.whT, nullptr, B.F_VH, N3, 32, 0);
    k_vn<<<gblk(N * 32), 256, 0, st>>>(B.F_VH, 32, B.F_VN, N, 32);
    k_concat_svn<<<gblk(Nr * 544), 256, 0, st>>>(B.F_MS, 512, 512, B.F_VN, 64, 32, B.BF_AS, 544, N, Nr * 544);
    gemm(st, B.BF_AS, f1.wswT, f1.wsb, B.F_MS, Nr, 128, 0);
    k_cvt_rows<<<gblk(N3 * 32), 256, 0, st>>>(B.F_VH, 32, B.BF_VH, 32, N3 * 32, 32);
    gemm(st, B.BF_VH, f1.wvT, nullptr, B.F_VO, N3, 16, 0);
    k_add<<<gblk(N * 128), 256, 0, st>>>(B.sCur, B.F_MS, N * 128);
    k_add<<<gblk(3 * N * 32), 256, 0, st>>>(B.Vp, B.F_VO, 3 * N * 32);
    k_ln_s<<<(N + 7) / 8, 256, 0, st>>>(B.sCur, B.sCur, w.L[l].n1.g, w.L[l].n1.b, N, 128, 128);
    k_ln_v<<<gblk(N), 256, 0, st>>>(B.Vp, 32, N, 16);
    // stash per-layer outputs for Wout concat
    k_copy_cols<<<gblk(N * 128), 256, 0, st>>>(B.sCur, 128, B.cS, 384, l * 128, 128, N * 128);
    k_copy_cols<<<gblk(3 * N * 16), 256, 0, st>>>(B.Vp, 32, B.cV, 64, l * 16, 16, 3 * N * 16);
  }
  // ---- Wout_ln + Wout (act=True, vo=0)
  k_ln_s<<<(N + 7) / 8, 256, 0, st>>>(B.cS, B.cS, w.Wout_ln.g, w.Wout_ln.b, N, 384, 384);
  k_ln_v<<<gblk(N), 256, 0, st>>>(B.cV, 64, N, 48);
  k_cvt_rows<<<gblk(N3 * 64), 256, 0, st>>>(B.cV, 64, B.BF_AV, 64, N3 * 64, 48);
  gemm(st, B.BF_AV, w.Wout.whT, nullptr, B.F_VH, N3, 48, 0);
  k_vn<<<gblk(N * 48), 256, 0, st>>>(B.F_VH, 64, B.F_VN, N, 48);
  k_concat_svn<<<gblk(Nr * 448), 256, 0, st>>>(B.cS, 384, 384, B.F_VN, 64, 48, B.BF_AS, 448, N, Nr * 448);
  gemm(st, B.BF_AS, w.Wout.wswT, w.Wout.wsb, B.F_MS, Nr, 384, 1);
  if (outDst)
    k_copy_cols<<<gblk(N * 384), 256, 0, st>>>(B.F_MS, 384, outDst, outLd, 0, 384, N * 384);
}

extern "C" void kernel_launch(void* const* d_in, const int* in_sizes, int n_in,
                              void* d_out, int out_size, void* d_ws, size_t ws_size,
                              hipStream_t stream) {
  (void)in_sizes; (void)out_size;
  const int NP = 10000, NL = 10000, NC = 20000, EP = 160000, EL = 160000, EC = 320000;
  const float* p_ns = (const float*)d_in[0];
  const float* p_nv = (const float*)d_in[1];
  const float* p_es = (const float*)d_in[2];
  const float* p_ev = (const float*)d_in[3];
  const int*   p_sq = (const int*)d_in[4];
  const int*   p_sr = (const int*)d_in[5];
  const int*   p_ds = (const int*)d_in[6];
  const float* l_ns = (const float*)d_in[7];
  const float* l_nv = (const float*)d_in[8];
  const float* l_es = (const float*)d_in[9];
  const float* l_ev = (const float*)d_in[10];
  const int*   l_sq = (const int*)d_in[11];
  const int*   l_sr = (const int*)d_in[12];
  const int*   l_ds = (const int*)d_in[13];
  const float* c_nv = (const float*)d_in[14];
  const float* c_es = (const float*)d_in[15];
  const float* c_ev = (const float*)d_in[16];
  const int*   c_sr = (const int*)d_in[17];
  const int*   c_ds = (const int*)d_in[18];
  const int*   n2g  = (const int*)d_in[19];

  // params: jax tree order (sorted dict keys): complex, dense, ligand, protein
  Rd rd; rd.din = d_in; rd.i = 20; rd.off = 0;
  rd.blobMode = (n_in <= 24);
  rd.blob = rd.blobMode ? (const float*)d_in[20] : nullptr;
  StageW Wc, Wl, Wp; DenseW Dn;
  rdStage(rd, Wc, 384, false);
  Dn.b1 = rd.take(768); Dn.b2 = rd.take(1);
  Dn.w1 = rd.take((size_t)384 * 768); Dn.w2 = rd.take(768);
  rdStage(rd, Wl, 26, true);
  rdStage(rd, Wp, 26, true);

  // workspace
  Arena ar{ (char*)d_ws, 0, ws_size };
  Bufs B;
  B.sCur = allocF(ar, (size_t)20000 * 128);
  B.Vp   = allocF(ar, (size_t)60000 * 32);
  B.sAcc = allocF(ar, (size_t)20000 * 128);
  B.VAcc = allocF(ar, (size_t)60000 * 32);
  B.cS   = allocF(ar, (size_t)20000 * 384);
  B.cV   = allocF(ar, (size_t)60000 * 64);
  B.esP  = allocF(ar, (size_t)320000 * 32);
  B.eVo  = allocF(ar, (size_t)320000 * 3);
  B.vnWe = allocF(ar, (size_t)320000);
  B.deg  = allocF(ar, (size_t)20000);
  B.s0p  = allocF(ar, (size_t)10000 * 32);
  B.s0c  = allocF(ar, (size_t)20000 * 416);
  B.F_VH = allocF(ar, (size_t)98304 * 64);
  B.F_VO = allocF(ar, (size_t)98304 * 32);
  B.F_VN = allocF(ar, (size_t)32768 * 64);
  B.F_MS = allocF(ar, (size_t)20000 * 768);
  B.gAcc = allocF(ar, 16);
  B.BF_AV = allocB(ar, (size_t)98304 * 64);
  B.BF_VH = allocB(ar, (size_t)98304 * 64);
  B.BF_AS = allocB(ar, (size_t)320000 * 64);
  if (ar.off + (size_t)16 * 1024 * 1024 > ws_size) return;  // insufficient scratch

  // weights -> transposed bf16 (recomputed every call; deterministic)
  prepStage(stream, ar, Wp);
  prepStage(stream, ar, Wl);
  prepStage(stream, ar, Wc);
  WMat w1T = prepW(stream, ar, Dn.w1, 384, 768);

  // complex-stage scalar input [NC, 416]: protein rows then ligand rows
  (void)hipMemsetAsync(B.s0c, 0, (size_t)20000 * 416 * 4, stream);
  runStage(stream, Wp, NP, EP, p_ns, p_nv, p_es, p_ev, p_sq, p_sr, p_ds,
           26, B.s0p, 32, B.s0c, 416, B);
  runStage(stream, Wl, NL, EL, l_ns, l_nv, l_es, l_ev, l_sq, l_sr, l_ds,
           26, B.s0p, 32, B.s0c + (size_t)10000 * 416, 416, B);
  runStage(stream, Wc, NC, EC, nullptr, c_nv, c_es, c_ev, nullptr, c_sr, c_ds,
           384, B.s0c, 416, nullptr, 0, B);

  // dense head: complex output lives in B.F_MS [NC, 384]
  float* out = (float*)d_out;
  k_cvt_rows<<<gblk(NC * 384), 256, 0, stream>>>(B.F_MS, 384, B.BF_AS, 384, NC * 384, 384);
  gemm(stream, B.BF_AS, w1T, Dn.b1, B.F_MS, NC, 768, 1);
  k_dot_head<<<(NC + 7) / 8, 256, 0, stream>>>(B.F_MS, Dn.w2, Dn.b2, out, NC, 768);
  (void)hipMemsetAsync(B.gAcc, 0, 16 * 4, stream);
  k_seg_graph<<<gblk(NC), 256, 0, stream>>>(out, n2g, B.gAcc, B.gAcc + 8, NC);
  k_gfinal<<<1, 8, 0, stream>>>(B.gAcc, B.gAcc + 8, out + NC);
}
